// CrossModalObjectiveAttentionBlock_78838419685629
// MI455X (gfx1250) — compile-verified, matrix operand order checked by reading, not run on hardware
//
#include <hip/hip_runtime.h>
#include <hip/hip_bf16.h>

typedef __bf16 bf16_t;
typedef __attribute__((ext_vector_type(16))) __bf16 v16bf;
typedef __attribute__((ext_vector_type(8)))  float  v8f;

#define Bz 16
#define Cc 256
#define HW 4096
#define NHd 4
#define HD 64
#define NT 4
#define GRP 32
#define COMB 512
#define EPSV 1e-5f

// LDS panel geometry: 64 rows x 512B data, padded to 528B row stride
#define PANEL_ROWB 512
#define PANEL_LDSB 528

__device__ __forceinline__ float bf2f(unsigned short u) {
    union { unsigned int i; float f; } x; x.i = ((unsigned int)u) << 16; return x.f;
}
__device__ __forceinline__ unsigned short f2bf(float f) {
    union { bf16_t h; unsigned short u; } x; x.h = (bf16_t)f; return x.u;
}

union FragU { uint4 u[2]; v16bf v; };

// ---------------------------------------------------------------------------
// Typed address-space helpers for the CDNA5 async-to-LDS builtin, whose
// signature is (int4 AS1*, int4 AS3*, imm offset, imm cpol).
// Generic LDS pointers carry the LDS offset in the low 32 bits; generic
// global pointers are the 64-bit VA.
// ---------------------------------------------------------------------------
typedef int v4i_t __attribute__((vector_size(16)));
typedef __attribute__((address_space(1))) v4i_t* as1_v4i;
typedef __attribute__((address_space(3))) v4i_t* as3_v4i;

__device__ __forceinline__ as1_v4i as_global(const void* p) {
    return reinterpret_cast<as1_v4i>(reinterpret_cast<uintptr_t>(p));
}
__device__ __forceinline__ as3_v4i as_lds(void* p) {
    return reinterpret_cast<as3_v4i>(
        (unsigned int)reinterpret_cast<uintptr_t>(p));
}

// Stage one contiguous 32KB global panel (64 rows x 512B) into LDS with
// 528B row pitch. Async DMA path when available (ASYNCcnt), else direct.
__device__ __forceinline__ void stage_panel(const unsigned char* __restrict__ gsrc,
                                            unsigned char* __restrict__ lds, int t) {
#pragma unroll
    for (int i = 0; i < 8; ++i) {
        int idx = t + i * 256;            // 2048 16B chunks
        int row = idx >> 5, ci = idx & 31;
#if __has_builtin(__builtin_amdgcn_global_load_async_to_lds_b128)
        __builtin_amdgcn_global_load_async_to_lds_b128(
            as_global(gsrc + row * PANEL_ROWB + ci * 16),
            as_lds(lds + row * PANEL_LDSB + ci * 16), 0, 0);
#else
        *(uint4*)(lds + row * PANEL_LDSB + ci * 16) =
            *(const uint4*)(gsrc + row * PANEL_ROWB + ci * 16);
#endif
    }
}

__device__ __forceinline__ void stage_wait() {
#if __has_builtin(__builtin_amdgcn_global_load_async_to_lds_b128)
    asm volatile("s_wait_asynccnt 0x0" ::: "memory");
#endif
    __syncthreads();
}

// WMMA fragment from a staged LDS panel. 16-bit 16x32 layout (ISA 7.12.2):
// lane<16 -> kb=0, lane>=16 -> kb=8; per lane two contiguous 16B runs at
// K = kb..kb+7 and kb+16..kb+23.
__device__ __forceinline__ v16bf lds_frag(const unsigned char* base,
                                          int row, int k0, int lane) {
    int kb = (lane >> 4) ? 8 : 0;
    const unsigned char* p = base + row * PANEL_LDSB + (k0 + kb) * 2;
    FragU f;
    f.u[0] = *(const uint4*)(p);
    f.u[1] = *(const uint4*)(p + 32);
    return f.v;
}

// ---------------------------------------------------------------------------
// Kernel 1: GroupNorm statistics. One block per (b, g); reduce 8*4096 elems.
// ---------------------------------------------------------------------------
__global__ void k_gn_stats(const float* __restrict__ img, float* __restrict__ stats) {
    int bg = blockIdx.x;
    const float* base = img + (size_t)bg * 8 * HW;
    float s = 0.f, s2 = 0.f;
    for (int i = threadIdx.x; i < 8 * HW; i += 256) {
        float v = base[i];
        s += v; s2 += v * v;
    }
    __shared__ float r1[256], r2[256];
    r1[threadIdx.x] = s; r2[threadIdx.x] = s2;
    __syncthreads();
    for (int off = 128; off > 0; off >>= 1) {
        if (threadIdx.x < off) {
            r1[threadIdx.x] += r1[threadIdx.x + off];
            r2[threadIdx.x] += r2[threadIdx.x + off];
        }
        __syncthreads();
    }
    if (threadIdx.x == 0) {
        float mean = r1[0] * (1.f / 32768.f);
        float var  = r2[0] * (1.f / 32768.f) - mean * mean;
        stats[bg * 2 + 0] = mean;
        stats[bg * 2 + 1] = rsqrtf(var + EPSV);
    }
}

// ---------------------------------------------------------------------------
// Kernel 2: apply GroupNorm affine, convert to bf16, transpose to [b][p][c].
// ---------------------------------------------------------------------------
__global__ void k_gn_apply(const float* __restrict__ img,
                           const float* __restrict__ stats,
                           const float* __restrict__ gn_w,
                           const float* __restrict__ gn_b,
                           unsigned short* __restrict__ xnT) {
    __shared__ unsigned short tile[64][65];
    int b  = blockIdx.z;
    int p0 = blockIdx.x * 64;
    int c0 = blockIdx.y * 64;
    int tx = threadIdx.x & 63;
    int ty = threadIdx.x >> 6;
#pragma unroll
    for (int i = 0; i < 16; ++i) {
        int cl = ty + i * 4;
        int c  = c0 + cl;
        int g  = c >> 3;
        float mean = stats[(b * GRP + g) * 2 + 0];
        float rstd = stats[(b * GRP + g) * 2 + 1];
        float v = img[((size_t)b * Cc + c) * HW + p0 + tx];
        float xn = (v - mean) * rstd * gn_w[c] + gn_b[c];
        tile[cl][tx] = f2bf(xn);
    }
    __syncthreads();
#pragma unroll
    for (int i = 0; i < 16; ++i) {
        int pl = ty + i * 4;
        xnT[((size_t)b * HW + p0 + pl) * Cc + c0 + tx] = tile[tx][pl];
    }
}

// ---------------------------------------------------------------------------
// Kernel 3: f32 -> bf16 weight conversion.
// ---------------------------------------------------------------------------
__global__ void k_cvt_bf16(const float* __restrict__ src, unsigned short* __restrict__ dst) {
    int i = blockIdx.x * 256 + threadIdx.x;
    dst[i] = f2bf(src[i]);
}

// ---------------------------------------------------------------------------
// Kernel 4: conditioning path: LayerNorm(combined) then kv = cn @ Wkv^T + bkv.
// ---------------------------------------------------------------------------
__global__ void k_cond_kv(const float* __restrict__ params,
                          const float* __restrict__ obj,
                          const float* __restrict__ ln_w,
                          const float* __restrict__ ln_b,
                          const float* __restrict__ Wkv,
                          const float* __restrict__ bkv,
                          float* __restrict__ kv) {
    int b  = blockIdx.x >> 3;
    int jc = blockIdx.x & 7;
    int t  = threadIdx.x;

    __shared__ float cn[COMB];
    __shared__ float r1[256], r2[256];
    __shared__ float mean_s, rstd_s;

    float v0 = (t < 128) ? params[b * 128 + t] : obj[b * 384 + (t - 128)];
    float v1 = obj[b * 384 + t + 128];
    r1[t] = v0 + v1;
    r2[t] = v0 * v0 + v1 * v1;
    __syncthreads();
    for (int off = 128; off > 0; off >>= 1) {
        if (t < off) { r1[t] += r1[t + off]; r2[t] += r2[t + off]; }
        __syncthreads();
    }
    if (t == 0) {
        float m = r1[0] * (1.f / 512.f);
        float var = r2[0] * (1.f / 512.f) - m * m;
        mean_s = m; rstd_s = rsqrtf(var + EPSV);
    }
    __syncthreads();
    float m = mean_s, rs = rstd_s;
    cn[t]       = (v0 - m) * rs * ln_w[t]       + ln_b[t];
    cn[t + 256] = (v1 - m) * rs * ln_w[t + 256] + ln_b[t + 256];
    __syncthreads();

    int j = jc * 256 + t;
    const float4* w4 = (const float4*)(Wkv + (size_t)j * COMB);
    const float4* c4 = (const float4*)cn;
    float acc = bkv[j];
#pragma unroll 4
    for (int i = 0; i < COMB / 4; ++i) {
        float4 w = w4[i], c = c4[i];
        acc += w.x * c.x + w.y * c.y + w.z * c.z + w.w * c.w;
    }
    kv[(size_t)b * 2048 + j] = acc;             // [b][2][256][4]
}

// ---------------------------------------------------------------------------
// Kernel 5: Q GEMM. q[b][o][p] = sum_c Wq[o][c] * xnT[b][p][c].
// Async-stage both 64x256 bf16 panels into LDS, feed WMMA from ds_load_b128.
// ---------------------------------------------------------------------------
__global__ void __launch_bounds__(256) k_gemm_q(const unsigned char* __restrict__ Wq,
                                                const unsigned char* __restrict__ xnT,
                                                unsigned short* __restrict__ qBf) {
    __shared__ __align__(16) unsigned char ldsA[64 * PANEL_LDSB];
    __shared__ __align__(16) unsigned char ldsB[64 * PANEL_LDSB];

    int t    = threadIdx.x;
    int lane = t & 31;
    int wave = t >> 5;
    int wm = wave & 3, wn = wave >> 2;
    int b = blockIdx.z;
    int Mbase = blockIdx.y * 64;
    int Nbase = blockIdx.x * 64;

    // Panels are contiguous 32KB blocks in our layouts.
    stage_panel(Wq + (size_t)Mbase * PANEL_ROWB, ldsA, t);
    stage_panel(xnT + ((size_t)b * HW + Nbase) * PANEL_ROWB, ldsB, t);
    stage_wait();

    int r = lane & 15;
    v8f acc0 = {0.f,0.f,0.f,0.f,0.f,0.f,0.f,0.f};
    v8f acc1 = {0.f,0.f,0.f,0.f,0.f,0.f,0.f,0.f};
#pragma unroll
    for (int k0 = 0; k0 < Cc; k0 += 32) {
        v16bf fa  = lds_frag(ldsA, wm * 16 + r, k0, lane);
        v16bf fb0 = lds_frag(ldsB, wn * 32 + r, k0, lane);
        v16bf fb1 = lds_frag(ldsB, wn * 32 + 16 + r, k0, lane);
        acc0 = __builtin_amdgcn_wmma_f32_16x16x32_bf16(false, fa, false, fb0,
                                                       (short)0, acc0, false, false);
        acc1 = __builtin_amdgcn_wmma_f32_16x16x32_bf16(false, fa, false, fb1,
                                                       (short)0, acc1, false, false);
    }
    int col = lane & 15;
    int rowAdd = (lane >> 4) * 8;
    unsigned short* Qb = qBf + (size_t)b * Cc * HW;
#pragma unroll
    for (int i = 0; i < 8; ++i) {
        int row = Mbase + wm * 16 + i + rowAdd;
        int nb  = Nbase + wn * 32;
        Qb[(size_t)row * HW + nb + col]      = f2bf(acc0[i]);
        Qb[(size_t)row * HW + nb + 16 + col] = f2bf(acc1[i]);
    }
}

// ---------------------------------------------------------------------------
// Kernel 6: attention over NT=4 tokens. Block = (b,head) x 512 pixels,
// 2 pixels/thread (dword q loads). Phase 2 writes hT packed bf16x2.
// ---------------------------------------------------------------------------
__global__ void k_attn(const unsigned short* __restrict__ qBf,
                       const float* __restrict__ kv,
                       unsigned short* __restrict__ hT) {
    int bh = blockIdx.y;
    int b = bh >> 2, head = bh & 3;
    int p0 = blockIdx.x * 512;
    int t = threadIdx.x;

    __shared__ float ks[HD][NT], vs[HD][NT];
    __shared__ float attn_s[512][NT];

    {
        int c = t >> 2, s = t & 3;
        ks[c][s] = kv[(size_t)b * 2048 +        (head * HD + c) * NT + s];
        vs[c][s] = kv[(size_t)b * 2048 + 1024 + (head * HD + c) * NT + s];
    }
    __syncthreads();

    // ---- phase 1: two pixels per thread
    const unsigned short* qbase =
        qBf + ((size_t)b * Cc + head * HD) * HW + p0 + 2 * t;
    float a00 = 0.f, a01 = 0.f, a02 = 0.f, a03 = 0.f;
    float a10 = 0.f, a11 = 0.f, a12 = 0.f, a13 = 0.f;
#pragma unroll 8
    for (int c = 0; c < HD; ++c) {
        unsigned int qq = *(const unsigned int*)(qbase + (size_t)c * HW);
        float q0 = bf2f((unsigned short)(qq & 0xffffu));
        float q1 = bf2f((unsigned short)(qq >> 16));
        float k0 = ks[c][0], k1 = ks[c][1], k2 = ks[c][2], k3 = ks[c][3];
        a00 += q0 * k0; a01 += q0 * k1; a02 += q0 * k2; a03 += q0 * k3;
        a10 += q1 * k0; a11 += q1 * k1; a12 += q1 * k2; a13 += q1 * k3;
    }
    const float sc = 1.0f / (float)HD;   // scale applied to both q and k -> 1/hd
    {
        float x0 = a00 * sc, x1 = a01 * sc, x2 = a02 * sc, x3 = a03 * sc;
        float mx = fmaxf(fmaxf(x0, x1), fmaxf(x2, x3));
        float e0 = __expf(x0 - mx), e1 = __expf(x1 - mx);
        float e2 = __expf(x2 - mx), e3 = __expf(x3 - mx);
        float inv = 1.f / (e0 + e1 + e2 + e3);
        attn_s[2 * t][0] = e0 * inv; attn_s[2 * t][1] = e1 * inv;
        attn_s[2 * t][2] = e2 * inv; attn_s[2 * t][3] = e3 * inv;
    }
    {
        float x0 = a10 * sc, x1 = a11 * sc, x2 = a12 * sc, x3 = a13 * sc;
        float mx = fmaxf(fmaxf(x0, x1), fmaxf(x2, x3));
        float e0 = __expf(x0 - mx), e1 = __expf(x1 - mx);
        float e2 = __expf(x2 - mx), e3 = __expf(x3 - mx);
        float inv = 1.f / (e0 + e1 + e2 + e3);
        attn_s[2 * t + 1][0] = e0 * inv; attn_s[2 * t + 1][1] = e1 * inv;
        attn_s[2 * t + 1][2] = e2 * inv; attn_s[2 * t + 1][3] = e3 * inv;
    }
    __syncthreads();

    // ---- phase 2: lane covers 2 channels -> contiguous 128B store per pixel
    int wave = t >> 5, lane = t & 31;
    int c = lane * 2;
    float v00 = vs[c][0],   v01 = vs[c][1],   v02 = vs[c][2],   v03 = vs[c][3];
    float v10 = vs[c+1][0], v11 = vs[c+1][1], v12 = vs[c+1][2], v13 = vs[c+1][3];
#pragma unroll 4
    for (int pi = 0; pi < 64; ++pi) {
        int pl = wave * 64 + pi;
        float w0 = attn_s[pl][0], w1 = attn_s[pl][1];
        float w2 = attn_s[pl][2], w3 = attn_s[pl][3];
        float h0 = w0 * v00 + w1 * v01 + w2 * v02 + w3 * v03;
        float h1 = w0 * v10 + w1 * v11 + w2 * v12 + w3 * v13;
        unsigned int pk = (unsigned int)f2bf(h0) | ((unsigned int)f2bf(h1) << 16);
        *(unsigned int*)&hT[((size_t)b * HW + p0 + pl) * Cc + head * HD + c] = pk;
    }
}

// ---------------------------------------------------------------------------
// Kernel 7: proj GEMM + bias + residual (LDS-staged, WMMA).
// out[b][o][p] = sum_c Wp[o][c] * hT[b][p][c] + bp[o] + img[b][o][p]
// ---------------------------------------------------------------------------
__global__ void __launch_bounds__(256) k_gemm_proj(const unsigned char* __restrict__ Wp,
                                                   const unsigned char* __restrict__ hT,
                                                   const float* __restrict__ bp,
                                                   const float* __restrict__ img,
                                                   float* __restrict__ out) {
    __shared__ __align__(16) unsigned char ldsA[64 * PANEL_LDSB];
    __shared__ __align__(16) unsigned char ldsB[64 * PANEL_LDSB];

    int t    = threadIdx.x;
    int lane = t & 31;
    int wave = t >> 5;
    int wm = wave & 3, wn = wave >> 2;
    int b = blockIdx.z;
    int Mbase = blockIdx.y * 64;
    int Nbase = blockIdx.x * 64;

    stage_panel(Wp + (size_t)Mbase * PANEL_ROWB, ldsA, t);
    stage_panel(hT + ((size_t)b * HW + Nbase) * PANEL_ROWB, ldsB, t);
    stage_wait();

    int r = lane & 15;
    v8f acc0 = {0.f,0.f,0.f,0.f,0.f,0.f,0.f,0.f};
    v8f acc1 = {0.f,0.f,0.f,0.f,0.f,0.f,0.f,0.f};
#pragma unroll
    for (int k0 = 0; k0 < Cc; k0 += 32) {
        v16bf fa  = lds_frag(ldsA, wm * 16 + r, k0, lane);
        v16bf fb0 = lds_frag(ldsB, wn * 32 + r, k0, lane);
        v16bf fb1 = lds_frag(ldsB, wn * 32 + 16 + r, k0, lane);
        acc0 = __builtin_amdgcn_wmma_f32_16x16x32_bf16(false, fa, false, fb0,
                                                       (short)0, acc0, false, false);
        acc1 = __builtin_amdgcn_wmma_f32_16x16x32_bf16(false, fa, false, fb1,
                                                       (short)0, acc1, false, false);
    }
    int col = lane & 15;
    int rowAdd = (lane >> 4) * 8;
#pragma unroll
    for (int i = 0; i < 8; ++i) {
        int row = Mbase + wm * 16 + i + rowAdd;
        int nb  = Nbase + wn * 32;
        float bias = bp[row];
        size_t i0 = ((size_t)b * Cc + row) * HW + nb + col;
        out[i0]      = acc0[i] + bias + img[i0];
        out[i0 + 16] = acc1[i] + bias + img[i0 + 16];
    }
}

// ---------------------------------------------------------------------------
// Host launcher
// ---------------------------------------------------------------------------
extern "C" void kernel_launch(void* const* d_in, const int* in_sizes, int n_in,
                              void* d_out, int out_size, void* d_ws, size_t ws_size,
                              hipStream_t stream) {
    (void)in_sizes; (void)n_in; (void)out_size; (void)ws_size;
    const float* img    = (const float*)d_in[0];
    const float* params = (const float*)d_in[1];
    const float* obj    = (const float*)d_in[2];
    const float* gn_w   = (const float*)d_in[3];
    const float* gn_b   = (const float*)d_in[4];
    const float* Wq     = (const float*)d_in[5];
    const float* ln_w   = (const float*)d_in[6];
    const float* ln_b   = (const float*)d_in[7];
    const float* Wkv    = (const float*)d_in[8];
    const float* bkv    = (const float*)d_in[9];
    const float* Wp     = (const float*)d_in[10];
    const float* bp     = (const float*)d_in[11];
    float* out = (float*)d_out;

    char* ws = (char*)d_ws;
    float*          stats = (float*)(ws + 0);                         //   4 KB
    float*          kvbuf = (float*)(ws + 4096);                      // 128 KB
    unsigned short* WqBf  = (unsigned short*)(ws + 135168);           // 128 KB
    unsigned short* WpBf  = (unsigned short*)(ws + 266240);           // 128 KB
    unsigned short* xnT   = (unsigned short*)(ws + 397312);           //  32 MB
    unsigned short* qBf   = (unsigned short*)(ws + 33951744ULL);      //  32 MB
    unsigned short* hT    = (unsigned short*)(ws + 67506176ULL);      //  32 MB

    k_gn_stats<<<dim3(Bz * GRP), dim3(256), 0, stream>>>(img, stats);
    k_gn_apply<<<dim3(HW / 64, Cc / 64, Bz), dim3(256), 0, stream>>>(
        img, stats, gn_w, gn_b, xnT);
    k_cvt_bf16<<<dim3(Cc * Cc / 256), dim3(256), 0, stream>>>(Wq, WqBf);
    k_cvt_bf16<<<dim3(Cc * Cc / 256), dim3(256), 0, stream>>>(Wp, WpBf);
    k_cond_kv<<<dim3(Bz * 8), dim3(256), 0, stream>>>(
        params, obj, ln_w, ln_b, Wkv, bkv, kvbuf);
    k_gemm_q<<<dim3(HW / 64, Cc / 64, Bz), dim3(256), 0, stream>>>(
        (const unsigned char*)WqBf, (const unsigned char*)xnT, qBf);
    k_attn<<<dim3(HW / 512, Bz * NHd), dim3(256), 0, stream>>>(qBf, kvbuf, hT);
    k_gemm_proj<<<dim3(HW / 64, Cc / 64, Bz), dim3(256), 0, stream>>>(
        (const unsigned char*)WpBf, (const unsigned char*)hT, bp, img, out);
}
